// SimpleDriftingLoss_3006477107513
// MI455X (gfx1250) — matrix operand order repaired and checked by reading, not compiled, not run on hardware
//
#include <hip/hip_runtime.h>
#include <hip/hip_bf16.h>

// ---------------------------------------------------------------------------
// SimpleDriftingLoss on MI455X (gfx1250).
// Matrix-core bound problem: two distance GEMMs + two softmax-weighted GEMMs
// (each 8192x8192x1024) on v_wmma_f32_16x16x32_bf16, wave32, 8 waves/block.
// Distance dots are stored bf16 in-place in the weight buffer so each GEMM is
// computed exactly once (548 GF total matrix work). Fast hardware sqrt/exp in
// the softmax epilogues so VALU co-executes cleanly against the WMMA pipe.
// ---------------------------------------------------------------------------

#define B_ROWS 8192
#define D_DIM  1024
#define INV_T  10.0f   // 1/temperature
#define NSLICE 8       // key-dim slices for the stats pass (occupancy)

typedef __attribute__((ext_vector_type(16))) __bf16 v16bf;
typedef __attribute__((ext_vector_type(8)))  float  v8f;
typedef __bf16 bf16;

union FragBF {
  v16bf  v;
  float4 q[2];
};

__device__ inline bf16 f2bf(float x) {
  unsigned u = __builtin_bit_cast(unsigned, x);
  u += 0x7FFFu + ((u >> 16) & 1u);          // round-to-nearest-even
  unsigned short h = (unsigned short)(u >> 16);
  return __builtin_bit_cast(bf16, h);
}

// Fast hardware sqrt (raw v_sqrt_f32, no IEEE refinement sequence).
__device__ inline float fast_sqrtf(float x) { return __builtin_amdgcn_sqrtf(x); }

// A fragment (16x32 bf16, K-major source row): lane L holds row M=L%16,
// elements 0-7 = K[8h..8h+8), elements 8-15 = K[16+8h..16+8h+8)  (h = L/16)
__device__ inline v16bf load_frag_a(const bf16* rowk, int lh) {
  FragBF f;
  f.q[0] = *(const float4*)(rowk + lh * 8);
  f.q[1] = *(const float4*)(rowk + lh * 8 + 16);
  return f.v;
}
// B fragment (32x16 bf16, K-major source row): lane L holds col N=L%16,
// elements 0-7 = K[8h..8h+8), elements 8-15 = K[16+8h..16+8h+8)  (h = L/16)
__device__ inline v16bf load_frag_b(const bf16* rowk, int lh) {
  FragBF f;
  f.q[0] = *(const float4*)(rowk + lh * 8);
  f.q[1] = *(const float4*)(rowk + lh * 8 + 16);
  return f.v;
}

__device__ inline float bfly_max16(float v) {
  v = fmaxf(v, __shfl_xor(v, 1));
  v = fmaxf(v, __shfl_xor(v, 2));
  v = fmaxf(v, __shfl_xor(v, 4));
  v = fmaxf(v, __shfl_xor(v, 8));
  return v;
}
__device__ inline float bfly_sum16(float v) {
  v += __shfl_xor(v, 1);
  v += __shfl_xor(v, 2);
  v += __shfl_xor(v, 4);
  v += __shfl_xor(v, 8);
  return v;
}

// ---------------------------------------------------------------------------
// Pass 1a: f32 -> bf16 row-major + row squared norms
// ---------------------------------------------------------------------------
__global__ __launch_bounds__(256) void k_convert(const float* __restrict__ src,
                                                 bf16* __restrict__ dst,
                                                 float* __restrict__ rnorm) {
  __shared__ float red[256];
  const int row = blockIdx.x;
  const float* s = src + (size_t)row * D_DIM;
  bf16* d = dst + (size_t)row * D_DIM;
  float a = 0.f;
  for (int c = threadIdx.x; c < D_DIM; c += 256) {
    float v = s[c];
    a += v * v;
    d[c] = f2bf(v);
  }
  red[threadIdx.x] = a;
  __syncthreads();
  for (int st = 128; st > 0; st >>= 1) {
    if (threadIdx.x < st) red[threadIdx.x] += red[threadIdx.x + st];
    __syncthreads();
  }
  if (threadIdx.x == 0) rnorm[row] = red[0];
}

// ---------------------------------------------------------------------------
// Pass 1b: bf16 RxC -> bf16 CxR (LDS tiled transpose, 64x64)
// ---------------------------------------------------------------------------
__global__ __launch_bounds__(256) void k_transpose(const bf16* __restrict__ src,
                                                   bf16* __restrict__ dst,
                                                   int R, int C) {
  __shared__ bf16 tile[64][65];
  const int rb = blockIdx.x * 64;
  const int cb = blockIdx.y * 64;
  const int tx = threadIdx.x & 63;
  const int ty0 = threadIdx.x >> 6;   // 0..3
  #pragma unroll
  for (int i = 0; i < 16; ++i) {
    int ty = ty0 * 16 + i;
    tile[ty][tx] = src[(size_t)(rb + ty) * C + cb + tx];
  }
  __syncthreads();
  #pragma unroll
  for (int i = 0; i < 16; ++i) {
    int ty = ty0 * 16 + i;
    dst[(size_t)(cb + ty) * R + rb + tx] = tile[tx][ty];
  }
}

// ---------------------------------------------------------------------------
// Pass 2: distance GEMM + softmax stats.
// grid = (rowBlk 0..63, slice 0..NSLICE-1). Each block: 128 rows x 1024 keys.
// Emits: raw dot products (bf16, into W) + per-slice partial (m, l).
// ---------------------------------------------------------------------------
template <bool MASK>
__global__ __launch_bounds__(256) void k_stats(const bf16* __restrict__ Q,
                                               const bf16* __restrict__ Km,
                                               const float* __restrict__ qn,
                                               const float* __restrict__ kn,
                                               float* __restrict__ pm_out,
                                               float* __restrict__ pl_out,
                                               bf16* __restrict__ Wdots) {
  __shared__ float qn_s[128], kn_s[128];
  __shared__ float pm[2][128], pl[2][128];
  __shared__ float run_m[128], run_l[128];

  const int tid = threadIdx.x;
  const int lane = tid & 31, wave = tid >> 5;
  const int ln = lane & 15, lh = lane >> 4;
  const int wr = wave >> 1, wc = wave & 1;
  const int rowBlk = blockIdx.x;
  const int keyBase = blockIdx.y * (B_ROWS / NSLICE);
  const float NEG_INF = -__builtin_inff();

  if (tid < 128) {
    qn_s[tid] = qn[rowBlk * 128 + tid];
    run_m[tid] = NEG_INF;
    run_l[tid] = 0.f;
  }
  __syncthreads();

  const bf16* arow0 = Q + (size_t)(rowBlk * 128 + wr * 32 + ln) * D_DIM;

  for (int jb = 0; jb < (B_ROWS / NSLICE) / 128; ++jb) {
    const int key0 = keyBase + jb * 128;
    if (tid < 128) kn_s[tid] = kn[key0 + tid];
    __syncthreads();

    v8f zero = {};
    v8f acc[2][4];
    #pragma unroll
    for (int rt = 0; rt < 2; ++rt)
      #pragma unroll
      for (int ct = 0; ct < 4; ++ct) acc[rt][ct] = zero;

    const bf16* brow0 = Km + (size_t)(key0 + wc * 64 + ln) * D_DIM;
    for (int kc = 0; kc < D_DIM; kc += 32) {
      v16bf af[2], bfr[4];
      #pragma unroll
      for (int rt = 0; rt < 2; ++rt)
        af[rt] = load_frag_a(arow0 + (size_t)(rt * 16) * D_DIM + kc, lh);
      #pragma unroll
      for (int ct = 0; ct < 4; ++ct)
        bfr[ct] = load_frag_b(brow0 + (size_t)(ct * 16) * D_DIM + kc, lh);
      #pragma unroll
      for (int rt = 0; rt < 2; ++rt)
        #pragma unroll
        for (int ct = 0; ct < 4; ++ct)
          acc[rt][ct] = __builtin_amdgcn_wmma_f32_16x16x32_bf16(
              false, af[rt], false, bfr[ct], (short)0, acc[rt][ct], false, false);
    }

    #pragma unroll
    for (int rt = 0; rt < 2; ++rt) {
      #pragma unroll
      for (int e = 0; e < 8; ++e) {
        const int li = wr * 32 + rt * 16 + e + 8 * lh;
        const int gi = rowBlk * 128 + li;
        float lg[4];
        #pragma unroll
        for (int ct = 0; ct < 4; ++ct) {
          const int lj = wc * 64 + ct * 16 + ln;
          const int gj = key0 + lj;
          const float dot = acc[rt][ct][e];
          Wdots[(size_t)gi * B_ROWS + gj] = f2bf(dot);   // raw dot, reused later
          float d2 = qn_s[li] + kn_s[lj] - 2.f * dot;
          float logit = -INV_T * fast_sqrtf(fmaxf(d2, 0.f));
          if (MASK && gi == gj) logit = NEG_INF;
          lg[ct] = logit;
        }
        float bm = bfly_max16(fmaxf(fmaxf(lg[0], lg[1]), fmaxf(lg[2], lg[3])));
        float s = __expf(lg[0] - bm) + __expf(lg[1] - bm) +
                  __expf(lg[2] - bm) + __expf(lg[3] - bm);
        float bl = bfly_sum16(s);
        if (ln == 0) { pm[wc][li] = bm; pl[wc][li] = bl; }
      }
    }
    __syncthreads();
    if (tid < 128) {
      float m0 = pm[0][tid], m1 = pm[1][tid];
      float l0 = pl[0][tid], l1 = pl[1][tid];
      float M = fmaxf(m0, m1);
      float L = l0 * __expf(m0 - M) + l1 * __expf(m1 - M);
      float rm = run_m[tid], rl = run_l[tid];
      float nm = fmaxf(rm, M);
      run_l[tid] = rl * __expf(rm - nm) + L * __expf(M - nm);
      run_m[tid] = nm;
    }
    __syncthreads();
  }
  if (tid < 128) {
    pm_out[(size_t)blockIdx.y * B_ROWS + rowBlk * 128 + tid] = run_m[tid];
    pl_out[(size_t)blockIdx.y * B_ROWS + rowBlk * 128 + tid] = run_l[tid];
  }
}

// ---------------------------------------------------------------------------
// Pass 2b: merge the NSLICE partial (m, l) pairs per row.
// ---------------------------------------------------------------------------
__global__ __launch_bounds__(256) void k_mergestats(const float* __restrict__ pm,
                                                    const float* __restrict__ pl,
                                                    float* __restrict__ m_out,
                                                    float* __restrict__ l_out) {
  const int i = blockIdx.x * 256 + threadIdx.x;
  float m = -__builtin_inff();
  #pragma unroll
  for (int s = 0; s < NSLICE; ++s) m = fmaxf(m, pm[(size_t)s * B_ROWS + i]);
  float l = 0.f;
  #pragma unroll
  for (int s = 0; s < NSLICE; ++s)
    l += pl[(size_t)s * B_ROWS + i] * __expf(pm[(size_t)s * B_ROWS + i] - m);
  m_out[i] = m;
  l_out[i] = l;
}

// ---------------------------------------------------------------------------
// Pass 3: in-place elementwise transform of W: bf16 dot -> normalized softmax
// weight. One block per row (coalesced, L2-resident stream).
// ---------------------------------------------------------------------------
template <bool MASK>
__global__ __launch_bounds__(256) void k_wfromdot(bf16* __restrict__ W,
                                                  const float* __restrict__ qn,
                                                  const float* __restrict__ kn,
                                                  const float* __restrict__ m_in,
                                                  const float* __restrict__ l_in) {
  const int gi = blockIdx.x;
  const float qni = qn[gi];
  const float mi = m_in[gi];
  const float liv = 1.f / l_in[gi];
  bf16* wrow = W + (size_t)gi * B_ROWS;
  for (int j = threadIdx.x; j < B_ROWS; j += 256) {
    float dot = (float)wrow[j];
    float d2 = qni + kn[j] - 2.f * dot;
    float logit = -INV_T * fast_sqrtf(fmaxf(d2, 0.f));
    float w = (MASK && gi == j) ? 0.f : __expf(logit - mi) * liv;
    wrow[j] = f2bf(w);
  }
}

// ---------------------------------------------------------------------------
// Pass 4: O = P @ V  (NT via pre-transposed Vt), fused drift epilogue.
// grid = (rowBlk, dBlk). FINAL=0: attraction pass. FINAL=1: repulsion+drift.
// ---------------------------------------------------------------------------
template <int FINAL>
__global__ __launch_bounds__(256) void k_apply(const bf16* __restrict__ W,
                                               const bf16* __restrict__ Vt,
                                               const float* __restrict__ genf,
                                               float* __restrict__ attr,
                                               float* __restrict__ rowA,
                                               float* __restrict__ rowR,
                                               float* __restrict__ rowD) {
  const int tid = threadIdx.x;
  const int lane = tid & 31, wave = tid >> 5;
  const int ln = lane & 15, lh = lane >> 4;
  const int wr = wave >> 1, wc = wave & 1;
  const int rowBlk = blockIdx.x, dBlk = blockIdx.y;

  v8f zero = {};
  v8f acc[2][4];
  #pragma unroll
  for (int rt = 0; rt < 2; ++rt)
    #pragma unroll
    for (int ct = 0; ct < 4; ++ct) acc[rt][ct] = zero;

  const bf16* arow0 = W + (size_t)(rowBlk * 128 + wr * 32 + ln) * B_ROWS;
  const bf16* brow0 = Vt + (size_t)(dBlk * 128 + wc * 64 + ln) * B_ROWS;
  for (int jc = 0; jc < B_ROWS; jc += 32) {
    v16bf af[2], bfr[4];
    #pragma unroll
    for (int rt = 0; rt < 2; ++rt)
      af[rt] = load_frag_a(arow0 + (size_t)(rt * 16) * B_ROWS + jc, lh);
    #pragma unroll
    for (int ct = 0; ct < 4; ++ct)
      bfr[ct] = load_frag_b(brow0 + (size_t)(ct * 16) * B_ROWS + jc, lh);
    #pragma unroll
    for (int rt = 0; rt < 2; ++rt)
      #pragma unroll
      for (int ct = 0; ct < 4; ++ct)
        acc[rt][ct] = __builtin_amdgcn_wmma_f32_16x16x32_bf16(
            false, af[rt], false, bfr[ct], (short)0, acc[rt][ct], false, false);
  }

  #pragma unroll
  for (int rt = 0; rt < 2; ++rt) {
    #pragma unroll
    for (int e = 0; e < 8; ++e) {
      const int li = wr * 32 + rt * 16 + e + 8 * lh;
      const int gi = rowBlk * 128 + li;
      float sA = 0.f, sR = 0.f, sD = 0.f;
      #pragma unroll
      for (int ct = 0; ct < 4; ++ct) {
        const int d = dBlk * 128 + wc * 64 + ct * 16 + ln;
        const size_t idx = (size_t)gi * D_DIM + d;
        float o = acc[rt][ct][e];
        float g = genf[idx];
        if (FINAL == 0) {
          float a = o - g;
          attr[idx] = a;
          sA += a * a;
        } else {
          float r = o - g;
          float a = attr[idx];
          float dr = a - r;
          sR += r * r;
          sD += dr * dr;
        }
      }
      if (FINAL == 0) {
        float t = bfly_sum16(sA);
        if (ln == 0) atomicAdd(&rowA[gi], t);
      } else {
        float tR = bfly_sum16(sR);
        float tD = bfly_sum16(sD);
        if (ln == 0) {
          atomicAdd(&rowR[gi], tR);
          atomicAdd(&rowD[gi], tD);
        }
      }
    }
  }
}

// ---------------------------------------------------------------------------
// Pass 5: final scalar reduction -> (loss, drift_norm, attraction_norm, rep_norm)
// ---------------------------------------------------------------------------
__global__ __launch_bounds__(256) void k_finalize(const float* __restrict__ rowA,
                                                  const float* __restrict__ rowR,
                                                  const float* __restrict__ rowD,
                                                  float* __restrict__ out) {
  __shared__ float s0[256], s1[256], s2[256], s3[256];
  const int tid = threadIdx.x;
  float la = 0.f, lr = 0.f, ld = 0.f, ls = 0.f;
  for (int i = tid; i < B_ROWS; i += 256) {
    float a = rowA[i], r = rowR[i], d = rowD[i];
    la += fast_sqrtf(fmaxf(a, 0.f));
    lr += fast_sqrtf(fmaxf(r, 0.f));
    ld += fast_sqrtf(fmaxf(d, 0.f));
    ls += d;
  }
  s0[tid] = la; s1[tid] = lr; s2[tid] = ld; s3[tid] = ls;
  __syncthreads();
  for (int st = 128; st > 0; st >>= 1) {
    if (tid < st) {
      s0[tid] += s0[tid + st];
      s1[tid] += s1[tid + st];
      s2[tid] += s2[tid + st];
      s3[tid] += s3[tid + st];
    }
    __syncthreads();
  }
  if (tid == 0) {
    out[0] = s3[0] / (float)((size_t)B_ROWS * D_DIM);  // loss = mean(drift^2)
    out[1] = s2[0] / (float)B_ROWS;                    // drift_norm
    out[2] = s0[0] / (float)B_ROWS;                    // attraction_norm
    out[3] = s1[0] / (float)B_ROWS;                    // repulsion_norm
  }
}

// ---------------------------------------------------------------------------
extern "C" void kernel_launch(void* const* d_in, const int* in_sizes, int n_in,
                              void* d_out, int out_size, void* d_ws, size_t ws_size,
                              hipStream_t stream) {
  (void)in_sizes; (void)n_in; (void)out_size; (void)ws_size;
  const float* gen = (const float*)d_in[0];
  const float* pos = (const float*)d_in[1];
  float* out = (float*)d_out;
  char* ws = (char*)d_ws;

  const size_t BF_MAT = (size_t)B_ROWS * D_DIM * sizeof(bf16);   // 16 MB
  const size_t F32_MAT = (size_t)B_ROWS * D_DIM * sizeof(float); // 32 MB
  const size_t W_BYTES = (size_t)B_ROWS * B_ROWS * sizeof(bf16); // 128 MB
  const size_t VEC = (size_t)B_ROWS * sizeof(float);             // 32 KB

  size_t off = 0;
  bf16* genb = (bf16*)(ws + off); off += BF_MAT;
  bf16* posb = (bf16*)(ws + off); off += BF_MAT;
  bf16* genT = (bf16*)(ws + off); off += BF_MAT;
  bf16* posT = (bf16*)(ws + off); off += BF_MAT;
  float* attr = (float*)(ws + off); off += F32_MAT;
  bf16* W = (bf16*)(ws + off); off += W_BYTES;
  float* gnorm = (float*)(ws + off); off += VEC;
  float* pnorm = (float*)(ws + off); off += VEC;
  float* m_a = (float*)(ws + off); off += VEC;
  float* l_a = (float*)(ws + off); off += VEC;
  float* m_r = (float*)(ws + off); off += VEC;
  float* l_r = (float*)(ws + off); off += VEC;
  float* pm = (float*)(ws + off); off += NSLICE * VEC;
  float* pl = (float*)(ws + off); off += NSLICE * VEC;
  float* rowAcc = (float*)(ws + off); off += 3 * VEC;  // rowA | rowR | rowD
  float* rowA = rowAcc;
  float* rowR = rowAcc + B_ROWS;
  float* rowD = rowAcc + 2 * B_ROWS;

  (void)hipMemsetAsync(rowAcc, 0, 3 * VEC, stream);

  // Precompute bf16 copies, transposes, row norms.
  k_convert<<<B_ROWS, 256, 0, stream>>>(gen, genb, gnorm);
  k_convert<<<B_ROWS, 256, 0, stream>>>(pos, posb, pnorm);
  k_transpose<<<dim3(B_ROWS / 64, D_DIM / 64), 256, 0, stream>>>(genb, genT, B_ROWS, D_DIM);
  k_transpose<<<dim3(B_ROWS / 64, D_DIM / 64), 256, 0, stream>>>(posb, posT, B_ROWS, D_DIM);

  // --- Attraction: softmax over distances to positives, O = P @ pos ---
  k_stats<false><<<dim3(B_ROWS / 128, NSLICE), 256, 0, stream>>>(
      genb, posb, gnorm, pnorm, pm, pl, W);
  k_mergestats<<<B_ROWS / 256, 256, 0, stream>>>(pm, pl, m_a, l_a);
  k_wfromdot<false><<<B_ROWS, 256, 0, stream>>>(W, gnorm, pnorm, m_a, l_a);
  k_apply<0><<<dim3(B_ROWS / 128, D_DIM / 128), 256, 0, stream>>>(
      W, posT, gen, attr, rowA, nullptr, nullptr);

  // --- Repulsion: softmax over distances to other generated (diag masked) ---
  k_stats<true><<<dim3(B_ROWS / 128, NSLICE), 256, 0, stream>>>(
      genb, genb, gnorm, gnorm, pm, pl, W);
  k_mergestats<<<B_ROWS / 256, 256, 0, stream>>>(pm, pl, m_r, l_r);
  k_wfromdot<true><<<B_ROWS, 256, 0, stream>>>(W, gnorm, gnorm, m_r, l_r);
  k_apply<1><<<dim3(B_ROWS / 128, D_DIM / 128), 256, 0, stream>>>(
      W, genT, gen, attr, nullptr, rowR, rowD);

  k_finalize<<<1, 256, 0, stream>>>(rowA, rowR, rowD, out);
}
